// SingleScaleGrid_65541200937503
// MI455X (gfx1250) — matrix operand order repaired and checked by reading, not compiled, not run on hardware
//
#include <hip/hip_runtime.h>
#include <hip/hip_bf16.h>

// ---------------- constants from the reference ----------------
#define RES        256
#define NRAYS      4096
#define NSAMP      446            // int(norm([257,257,257]))+1
#define TILES      28             // ceil(446/16)
#define SROW       144            // LDS activation row stride in halfs (16B-aligned rows)
#define WAVES      4
#define BLOCK      (WAVES * 32)
#define VOXEL      (2.0f / 256.0f)
#define NEARC      0.2f
#define FARC       3.0f
#define ACT_SHIFT  (-4.59511985f) // log(1/(1-0.01)-1)
#define CH_STRIDE  16777216       // RES^3

typedef __attribute__((ext_vector_type(16))) _Float16 v16h;
typedef __attribute__((ext_vector_type(8)))  _Float16 v8h;
typedef __attribute__((ext_vector_type(8)))  float    v8f;
typedef __attribute__((ext_vector_type(2)))  float    v2f;

// ---------------- weight prep: f32 [K][N] -> f16 [Npad][Kpad] (transposed, zero pad) ----
__global__ void prep_wt(const float* __restrict__ src, _Float16* __restrict__ dst,
                        int K, int N, int Kpad, int Npad) {
    int i = blockIdx.x * blockDim.x + threadIdx.x;
    if (i >= Kpad * Npad) return;
    int n = i / Kpad, k = i % Kpad;
    float v = (k < K && n < N) ? src[k * N + n] : 0.0f;
    dst[i] = (_Float16)v;
}

// mW1 [4][128] f32 -> [128][4] f32 (for the f32 16x16x4 WMMA)
__global__ void prep_w1(const float* __restrict__ src, float* __restrict__ dst) {
    int i = blockIdx.x * blockDim.x + threadIdx.x;
    if (i < 512) { int n = i >> 2, k = i & 3; dst[i] = src[k * 128 + n]; }
}

// ---------------- WMMA tile GEMM: 16xKPAD (LDS f16) @ KPADxNOUT (global f16^T) ----------
// nt loop deliberately NOT unrolled: keeps one v8f accumulator + one A/B fragment pair
// live at a time -> <=256 VGPRs -> 4 waves/SIMD to hide the trilinear gather latency.
template <int KPAD, int NOUT, bool RELU>
__device__ __forceinline__ void gemm_tile(const _Float16* src, const _Float16* wt,
                                          _Float16* dst, const float* bias, int nbias,
                                          int lane) {
    constexpr int NKB = KPAD / 32;
#pragma unroll 1
    for (int nt = 0; nt < NOUT / 16; ++nt) {
        v8f acc = {};
#pragma unroll
        for (int kb = 0; kb < NKB; ++kb) {
            // A fragment: lane m=lane&15, K = kb*32 + {0..7,16..23} (lanes 0-15) / {8..15,24..31}
            const _Float16* pa = src + (lane & 15) * SROW + kb * 32 + ((lane >> 4) << 3);
            v8h alo = *(const v8h*)pa;
            v8h ahi = *(const v8h*)(pa + 16);
            v16h A = __builtin_shufflevector(alo, ahi, 0,1,2,3,4,5,6,7,8,9,10,11,12,13,14,15);
            // B fragment: column n=lane&15 (+nt*16), K = kb*32 + {0..7,16..23} (lanes 0-15) / {8..15,24..31}
            const _Float16* pb = wt + (size_t)(nt * 16 + (lane & 15)) * KPAD + kb * 32 + ((lane >> 4) << 3);
            v8h blo = *(const v8h*)pb;
            v8h bhi = *(const v8h*)(pb + 16);
            v16h B = __builtin_shufflevector(blo, bhi, 0,1,2,3,4,5,6,7,8,9,10,11,12,13,14,15);
            acc = __builtin_amdgcn_wmma_f32_16x16x32_f16(false, A, false, B, (short)0, acc,
                                                         false, false);
        }
        int n = nt * 16 + (lane & 15);
        float bv = (bias && n < nbias) ? bias[n] : 0.0f;
        int mb = (lane >> 4) << 3;
#pragma unroll
        for (int r = 0; r < 8; ++r) {
            float v = acc[r] + bv;
            if (RELU) v = fmaxf(v, 0.0f);
            dst[(mb + r) * SROW + n] = (_Float16)v;
        }
    }
}

// ---------------- main kernel: one ray per wave32 -----------------
__global__ __launch_bounds__(BLOCK)
__attribute__((amdgpu_waves_per_eu(4)))           // cap ~256 VGPRs: no MSB addressing, 4 waves/SIMD
void nerf_march_kernel(
    const float* __restrict__ rays_o, const float* __restrict__ rays_d,
    const float* __restrict__ viewdirs, const float* __restrict__ grid,
    const float* __restrict__ wt1,                      // f32 [128][4]
    const _Float16* __restrict__ wt2, const _Float16* __restrict__ wt3,
    const _Float16* __restrict__ wtf,
    const _Float16* __restrict__ wtp1, const _Float16* __restrict__ wtp2,
    const _Float16* __restrict__ wtp3,
    const _Float16* __restrict__ wtd1, const _Float16* __restrict__ wtd2,
    const _Float16* __restrict__ wtd3, const _Float16* __restrict__ wtd4,
    const float* __restrict__ mb1, const float* __restrict__ mb2,
    const float* __restrict__ mb3, const float* __restrict__ pb1,
    const float* __restrict__ pb2, const float* __restrict__ pb3,
    const float* __restrict__ db1, const float* __restrict__ db2,
    const float* __restrict__ db3, const float* __restrict__ db4,
    float* __restrict__ out) {
    __shared__ __align__(16) _Float16 shA[WAVES][16 * SROW];
    __shared__ __align__(16) _Float16 shB[WAVES][16 * SROW];
    __shared__ __align__(16) _Float16 shC[WAVES][16 * SROW];
    __shared__ __align__(16) float    shMp[WAVES][16][4];

    const int lane = threadIdx.x & 31;
    const int wv = threadIdx.x >> 5;
    const int ray = blockIdx.x * WAVES + wv;

    _Float16* A = shA[wv];
    _Float16* Bf = shB[wv];
    _Float16* C = shC[wv];
    float(*MP)[4] = shMp[wv];

    // ---- per-ray setup (every lane redundantly) ----
    float o[3], d[3], vd[3];
#pragma unroll
    for (int c = 0; c < 3; ++c) {
        o[c] = rays_o[ray * 3 + c];
        d[c] = rays_d[ray * 3 + c];
        vd[c] = viewdirs[ray * 3 + c];
    }
    float tmn = -1e30f, tmx = 1e30f;
#pragma unroll
    for (int c = 0; c < 3; ++c) {
        float vc = (d[c] == 0.0f) ? 1e-6f : d[c];
        float ra = (1.0f - o[c]) / vc;
        float rb = (-1.0f - o[c]) / vc;
        tmn = fmaxf(tmn, fminf(ra, rb));
        tmx = fminf(tmx, fmaxf(ra, rb));
    }
    float tmin = fminf(fmaxf(tmn, NEARC), FARC);
    float tmax = fminf(fmaxf(tmx, NEARC), FARC);
    bool rayok = !(tmax <= tmin);
    float nd = sqrtf(d[0] * d[0] + d[1] * d[1] + d[2] * d[2]);

    // ---- direction embedding into C cols 15..53 (cols 54..63 zero), per sample row ----
    if (lane < 16) {
        _Float16* row = C + lane * SROW;
        row[15] = (_Float16)vd[0];
        row[16] = (_Float16)vd[1];
        row[17] = (_Float16)vd[2];
#pragma unroll 1
        for (int f = 0; f < 6; ++f) {
            float fr = (float)(1 << f);
#pragma unroll
            for (int j = 0; j < 3; ++j) {
                row[18 + f * 6 + j]     = (_Float16)sinf(vd[j] * fr);
                row[18 + f * 6 + 3 + j] = (_Float16)cosf(vd[j] * fr);
            }
        }
#pragma unroll
        for (int j = 54; j < 64; ++j) row[j] = (_Float16)0.0f;
    }

    float T = 1.0f;
    float acc0 = 0.0f, acc1 = 0.0f, acc2 = 0.0f;

    for (int t = 0; t < TILES; ++t) {
        int s = t * 16 + (lane & 15);
        float alpha = 0.0f;
        bool valid = false;

        // ---- phase 1: sample point + trilinear gather (lanes 0..15) ----
        if (lane < 16) {
            float ti = tmin + ((float)s * VOXEL) / nd;
            float p[3];
#pragma unroll
            for (int c = 0; c < 3; ++c) p[c] = o[c] + d[c] * ti;
            valid = rayok && (s < NSAMP) && p[0] >= -1.0f && p[0] <= 1.0f &&
                    p[1] >= -1.0f && p[1] <= 1.0f && p[2] >= -1.0f && p[2] <= 1.0f;
            int i0[3];
            float ff[3];
#pragma unroll
            for (int c = 0; c < 3; ++c) {
                float u = (p[c] + 1.0f) * 0.5f * 255.0f;
                u = fminf(fmaxf(u, 0.0f), 255.0f);
                int ii = (int)floorf(u);
                ii = ii < 254 ? ii : 254;
                ii = ii > 0 ? ii : 0;
                i0[c] = ii;
                ff[c] = u - (float)ii;
            }
            float mpv[4] = {0.0f, 0.0f, 0.0f, 0.0f};
            int bx = i0[0] * 65536, by = i0[1] * 256, bz = i0[2];
#pragma unroll
            for (int ci = 0; ci < 2; ++ci)
#pragma unroll
                for (int cj = 0; cj < 2; ++cj)
#pragma unroll
                    for (int ck = 0; ck < 2; ++ck) {
                        float w = (ci ? ff[0] : 1.0f - ff[0]) * (cj ? ff[1] : 1.0f - ff[1]) *
                                  (ck ? ff[2] : 1.0f - ff[2]);
                        int off = bx + ci * 65536 + by + cj * 256 + bz + ck;
#pragma unroll
                        for (int c4 = 0; c4 < 4; ++c4)
                            mpv[c4] += w * grid[c4 * CH_STRIDE + off];
                    }
#pragma unroll
            for (int c4 = 0; c4 < 4; ++c4) MP[lane][c4] = mpv[c4];
        }
        __syncthreads();

        // ---- layer m1: [16x4] @ [4x128] via V_WMMA_F32_16X16X4_F32, relu ----
        {
            int m = lane & 15;
            int k0 = (lane >> 4) * 2;
            v2f a;
            a[0] = MP[m][k0];
            a[1] = MP[m][k0 + 1];
            int mb = (lane >> 4) * 8;
#pragma unroll 1
            for (int nt = 0; nt < 8; ++nt) {
                int n = nt * 16 + (lane & 15);
                v2f b;
                b[0] = wt1[n * 4 + k0];
                b[1] = wt1[n * 4 + k0 + 1];
                v8f acc = {};
                acc = __builtin_amdgcn_wmma_f32_16x16x4_f32(false, a, false, b, (short)0, acc,
                                                            false, false);
                float bv = mb1[n];
#pragma unroll
                for (int r = 0; r < 8; ++r)
                    A[(mb + r) * SROW + n] = (_Float16)fmaxf(acc[r] + bv, 0.0f);
            }
        }
        __syncthreads();

        gemm_tile<128, 128, true>(A, wt2, Bf, mb2, 128, lane);  __syncthreads();
        gemm_tile<128, 32, false>(Bf, wt3, A, mb3, 32, lane);   __syncthreads();

        // ---- softmax over 32 features per sample (two-pass, no register array) ----
        if (lane < 16) {
            _Float16* row = A + lane * SROW;
            float mx = -1e30f;
#pragma unroll 1
            for (int j = 0; j < 32; ++j) mx = fmaxf(mx, (float)row[j]);
            float sum = 0.0f;
#pragma unroll 1
            for (int j = 0; j < 32; ++j) sum += expf((float)row[j] - mx);
            float inv = 1.0f / sum;
#pragma unroll 1
            for (int j = 0; j < 32; ++j)
                row[j] = (_Float16)(expf((float)row[j] - mx) * inv);
        }
        __syncthreads();

        gemm_tile<32, 64, false>(A, wtf, Bf, nullptr, 0, lane); __syncthreads();
        gemm_tile<64, 64, true>(Bf, wtp1, A, pb1, 64, lane);    __syncthreads();
        gemm_tile<64, 64, true>(A, wtp2, Bf, pb2, 64, lane);    __syncthreads();
        gemm_tile<64, 16, false>(Bf, wtp3, A, pb3, 16, lane);   __syncthreads();

        // ---- alpha + rgb latent ----
        if (lane < 16) {
            const _Float16* row = A + lane * SROW;
            float x = (float)row[0] + ACT_SHIFT;
            float sp = (x > 20.0f) ? x : log1pf(expf(x));
            alpha = valid ? (1.0f - expf(-sp)) : 0.0f;
            _Float16* crow = C + lane * SROW;
#pragma unroll
            for (int j = 0; j < 15; ++j)
                crow[j] = valid ? row[1 + j] : (_Float16)0.0f;
        }
        __syncthreads();

        gemm_tile<64, 64, true>(C, wtd1, A, db1, 64, lane);   __syncthreads();
        gemm_tile<64, 64, true>(A, wtd2, Bf, db2, 64, lane);  __syncthreads();
        gemm_tile<64, 64, true>(Bf, wtd3, A, db3, 64, lane);  __syncthreads();
        gemm_tile<64, 16, false>(A, wtd4, Bf, db4, 3, lane);  __syncthreads();

        // ---- sigmoid, transmittance scan, accumulate ----
        float r3[3] = {0.0f, 0.0f, 0.0f};
        if (lane < 16) {
            const _Float16* row = Bf + lane * SROW;
#pragma unroll
            for (int c = 0; c < 3; ++c) {
                float rv = (float)row[c];
                r3[c] = valid ? (1.0f / (1.0f + expf(-rv))) : 0.0f;
            }
        }
        float la = logf(fmaxf(1.0f - alpha, 1e-10f));  // 0 for lanes >= 16
        float incl = la;
#pragma unroll
        for (int off = 1; off < 16; off <<= 1) {
            float v = __shfl_up(incl, off, 32);
            if (lane >= off) incl += v;
        }
        float excl = incl - la;
        float w = alpha * T * expf(excl);
        acc0 += w * r3[0];
        acc1 += w * r3[1];
        acc2 += w * r3[2];
        float tot = __shfl(incl, 15, 32);
        T *= expf(tot);
        __syncthreads();
    }

    // ---- cross-lane reduce, lane 0 writes ----
#pragma unroll
    for (int off = 16; off >= 1; off >>= 1) {
        acc0 += __shfl_xor(acc0, off, 32);
        acc1 += __shfl_xor(acc1, off, 32);
        acc2 += __shfl_xor(acc2, off, 32);
    }
    if (lane == 0) {
        out[ray * 3 + 0] = acc0 + T;  // BG = 1.0
        out[ray * 3 + 1] = acc1 + T;
        out[ray * 3 + 2] = acc2 + T;
    }
}

// ---------------- workspace layout (bytes) ----------------
#define WT1_OFF 0        // f32 [128][4]   2048
#define WT2_OFF 2048     // f16 [128][128] 32768
#define WT3_OFF 34816    // f16 [32][128]  8192
#define WTF_OFF 43008    // f16 [64][32]   4096
#define WTP1_OFF 47104   // f16 [64][64]   8192
#define WTP2_OFF 55296
#define WTP3_OFF 63488   // f16 [16][64]   2048
#define WTD1_OFF 65536
#define WTD2_OFF 73728
#define WTD3_OFF 81920
#define WTD4_OFF 90112   // f16 [16][64]   2048  -> total 92160

extern "C" void kernel_launch(void* const* d_in, const int* in_sizes, int n_in,
                              void* d_out, int out_size, void* d_ws, size_t ws_size,
                              hipStream_t stream) {
    const float* rays_o = (const float*)d_in[0];
    const float* rays_d = (const float*)d_in[1];
    const float* viewdirs = (const float*)d_in[2];
    const float* grid = (const float*)d_in[3];
    const float* fv = (const float*)d_in[4];
    const float* mW1 = (const float*)d_in[5];  const float* mb1 = (const float*)d_in[6];
    const float* mW2 = (const float*)d_in[7];  const float* mb2 = (const float*)d_in[8];
    const float* mW3 = (const float*)d_in[9];  const float* mb3 = (const float*)d_in[10];
    const float* pW1 = (const float*)d_in[11]; const float* pb1 = (const float*)d_in[12];
    const float* pW2 = (const float*)d_in[13]; const float* pb2 = (const float*)d_in[14];
    const float* pW3 = (const float*)d_in[15]; const float* pb3 = (const float*)d_in[16];
    const float* dW1 = (const float*)d_in[17]; const float* db1 = (const float*)d_in[18];
    const float* dW2 = (const float*)d_in[19]; const float* db2 = (const float*)d_in[20];
    const float* dW3 = (const float*)d_in[21]; const float* db3 = (const float*)d_in[22];
    const float* dW4 = (const float*)d_in[23]; const float* db4 = (const float*)d_in[24];

    char* ws = (char*)d_ws;
    float* wt1 = (float*)(ws + WT1_OFF);
    _Float16* wt2 = (_Float16*)(ws + WT2_OFF);
    _Float16* wt3 = (_Float16*)(ws + WT3_OFF);
    _Float16* wtf = (_Float16*)(ws + WTF_OFF);
    _Float16* wtp1 = (_Float16*)(ws + WTP1_OFF);
    _Float16* wtp2 = (_Float16*)(ws + WTP2_OFF);
    _Float16* wtp3 = (_Float16*)(ws + WTP3_OFF);
    _Float16* wtd1 = (_Float16*)(ws + WTD1_OFF);
    _Float16* wtd2 = (_Float16*)(ws + WTD2_OFF);
    _Float16* wtd3 = (_Float16*)(ws + WTD3_OFF);
    _Float16* wtd4 = (_Float16*)(ws + WTD4_OFF);

    auto prep = [&](const float* src, _Float16* dst, int K, int N, int Kp, int Np) {
        int tot = Kp * Np;
        prep_wt<<<(tot + 255) / 256, 256, 0, stream>>>(src, dst, K, N, Kp, Np);
    };
    prep_w1<<<2, 256, 0, stream>>>(mW1, wt1);
    prep(mW2, wt2, 128, 128, 128, 128);
    prep(mW3, wt3, 128, 32, 128, 32);
    prep(fv,  wtf, 32, 64, 32, 64);
    prep(pW1, wtp1, 64, 64, 64, 64);
    prep(pW2, wtp2, 64, 64, 64, 64);
    prep(pW3, wtp3, 64, 16, 64, 16);
    prep(dW1, wtd1, 54, 64, 64, 64);
    prep(dW2, wtd2, 64, 64, 64, 64);
    prep(dW3, wtd3, 64, 64, 64, 64);
    prep(dW4, wtd4, 64, 3, 64, 16);

    nerf_march_kernel<<<NRAYS / WAVES, BLOCK, 0, stream>>>(
        rays_o, rays_d, viewdirs, grid, wt1, wt2, wt3, wtf, wtp1, wtp2, wtp3, wtd1,
        wtd2, wtd3, wtd4, mb1, mb2, mb3, pb1, pb2, pb3, db1, db2, db3, db4,
        (float*)d_out);
}
